// NAPLConvLayer_66632122630765
// MI455X (gfx1250) — matrix operand order repaired and checked by reading, not compiled, not run on hardware
//
#include <hip/hip_runtime.h>
#include <hip/hip_bf16.h>

// Problem constants (from reference): B,N,C,H,D,K
#define PB 64
#define PN 4096
#define PC 32
#define PH 64
#define PD 16
#define PK 3
#define NCOLS (PB * PC)   // 2048 columns (b,c) for the diffusion GEMM

typedef __bf16 v16bf __attribute__((ext_vector_type(16)));
typedef float  v8f   __attribute__((ext_vector_type(8)));
typedef int    v4i   __attribute__((ext_vector_type(4)));

union BfFrag {
    v16bf bf;
    v4i   q[2];
};

__device__ inline unsigned short f2bf(float f) {
    unsigned int u = __builtin_bit_cast(unsigned int, f);
    unsigned int r = u + 0x7FFFu + ((u >> 16) & 1u);   // round-to-nearest-even
    return (unsigned short)(r >> 16);
}

__device__ inline v8f v8f_zero() {
    v8f z;
#pragma unroll
    for (int i = 0; i < 8; ++i) z[i] = 0.0f;
    return z;
}

// ---------------------------------------------------------------------------
// Prep kernels (all trivially bandwidth-bound, tiny vs the main GEMM)
// ---------------------------------------------------------------------------

// As (K,N,N) fp32 -> bf16, same layout (100 MB, fits in 192 MB L2)
__global__ __launch_bounds__(256) void k_cvt_as(const float* __restrict__ As,
                                                unsigned short* __restrict__ Abf,
                                                long n) {
    long idx = (long)blockIdx.x * 256 + threadIdx.x;
    if (idx < n) Abf[idx] = f2bf(As[idx]);
}

// x (B,N,C) fp32 -> x_cm[(b*32+c)][m] bf16 (column-major view for B-fragments)
__global__ __launch_bounds__(256) void k_build_xcm(const float* __restrict__ x,
                                                   unsigned short* __restrict__ xcm) {
    long idx = (long)blockIdx.x * 256 + threadIdx.x;   // total NCOLS*PN = 8388608
    if (idx >= (long)NCOLS * PN) return;
    int m = (int)(idx & (PN - 1));
    int j = (int)(idx >> 12);
    int b = j >> 5;
    int c = j & 31;
    xcm[idx] = f2bf(x[((size_t)b * PN + m) * PC + c]);
}

// wT[n][o][k][i] = sum_d emb[n,d] * Wp[d,k,i,o]  (bf16; transposed so kernel-3
// B-fragments read contiguous K runs)
__global__ __launch_bounds__(256) void k_build_wt(const float* __restrict__ emb,
                                                  const float* __restrict__ Wp,
                                                  unsigned short* __restrict__ wT) {
    long idx = (long)blockIdx.x * 256 + threadIdx.x;   // total N*H*K*C = 25165824
    if (idx >= (long)PN * PH * PK * PC) return;
    int i = (int)(idx & 31);
    long t = idx >> 5;
    int k = (int)(t % PK);
    t /= PK;
    int o = (int)(t & (PH - 1));
    int n = (int)(t >> 6);
    float acc = 0.0f;
#pragma unroll
    for (int d = 0; d < PD; ++d)
        acc += emb[n * PD + d] * Wp[(((size_t)d * PK + k) * PC + i) * PH + o];
    wT[idx] = f2bf(acc);
}

// bias[n][o] = sum_d emb[n,d] * bp[d,o]  (fp32)
__global__ __launch_bounds__(256) void k_build_bias(const float* __restrict__ emb,
                                                    const float* __restrict__ bp,
                                                    float* __restrict__ bias) {
    long idx = (long)blockIdx.x * 256 + threadIdx.x;   // total N*H = 262144
    if (idx >= (long)PN * PH) return;
    int o = (int)(idx & (PH - 1));
    int n = (int)(idx >> 6);
    float acc = 0.0f;
#pragma unroll
    for (int d = 0; d < PD; ++d) acc += emb[n * PD + d] * bp[d * PH + o];
    bias[idx] = acc;
}

// ---------------------------------------------------------------------------
// Kernel 2: diffusion GEMM.  Per k:  Y[n, j=(b,c)] = sum_m A_k[n,m] * x_cm[j][m]
// Wave tile: 64 n x 64 j.  WG = 8 waves -> 128 n x 256 j.
// Output stored as x_g[n][b][k][i] bf16 (A-matrix layout for kernel 3).
// ---------------------------------------------------------------------------
__global__ __launch_bounds__(256) void k_diffusion(const unsigned short* __restrict__ Abf,
                                                   const unsigned short* __restrict__ xcm,
                                                   unsigned short* __restrict__ xg) {
    const int lane = threadIdx.x & 31;
    const int wave = threadIdx.x >> 5;
    const int half = lane >> 4;     // 0 or 1
    const int l16  = lane & 15;
    const int k    = blockIdx.z;
    const int n0   = blockIdx.x * 128 + (wave >> 2) * 64;
    const int j0   = blockIdx.y * 256 + (wave & 3) * 64;

    v8f acc[4][4];
#pragma unroll
    for (int r = 0; r < 4; ++r)
#pragma unroll
        for (int c = 0; c < 4; ++c) acc[r][c] = v8f_zero();

    const unsigned short* Abase = Abf + (size_t)k * PN * PN;

    for (int mi = 0; mi < PN; mi += 32) {
        BfFrag a[4], b[4];
        // A fragment (16x32 bf16): lane<16 holds row, K=[0..7],[16..23];
        // lane>=16 same row, K=[8..15],[24..31]  (ISA 7.12.2 16-bit A table)
#pragma unroll
        for (int r = 0; r < 4; ++r) {
            const unsigned short* p =
                Abase + (size_t)(n0 + r * 16 + l16) * PN + mi + half * 8;
            a[r].q[0] = *reinterpret_cast<const v4i*>(p);
            a[r].q[1] = *reinterpret_cast<const v4i*>(p + 16);
        }
        // B fragment (32x16 bf16): lane holds column j=l16, K=half*8+[0..7],[16..23]
        // contiguous -> x_cm rows are contiguous in m.
#pragma unroll
        for (int c = 0; c < 4; ++c) {
            const unsigned short* p =
                xcm + (size_t)(j0 + c * 16 + l16) * PN + mi + half * 8;
            b[c].q[0] = *reinterpret_cast<const v4i*>(p);
            b[c].q[1] = *reinterpret_cast<const v4i*>(p + 16);
        }
#pragma unroll
        for (int r = 0; r < 4; ++r)
#pragma unroll
            for (int c = 0; c < 4; ++c)
                acc[r][c] = __builtin_amdgcn_wmma_f32_16x16x32_bf16(
                    false, a[r].bf, false, b[c].bf, (short)0, acc[r][c],
                    false, false);
    }

    // Store: D layout row M = r + 8*half, col = l16. x_g[n][b][k][i] bf16.
#pragma unroll
    for (int rf = 0; rf < 4; ++rf) {
#pragma unroll
        for (int cf = 0; cf < 4; ++cf) {
            int j = j0 + cf * 16 + l16;
            int bb = j >> 5;
            int cc = j & 31;
#pragma unroll
            for (int r = 0; r < 8; ++r) {
                int n = n0 + rf * 16 + r + 8 * half;
                xg[(((size_t)n * PB + bb) * PK + k) * PC + cc] =
                    f2bf(acc[rf][cf][r]);
            }
        }
    }
}

// ---------------------------------------------------------------------------
// Kernel 3: per-node output GEMM.  For node n:
//   out[b, n, o] = sum_{k,i} x_g[n][b][k*32+i] * wT[n][o][k*32+i] + bias[n][o]
// One wave per n: 64b x 64o tile, K = 96 in 3 chunks of 32.
// ---------------------------------------------------------------------------
__global__ __launch_bounds__(128) void k_output(const unsigned short* __restrict__ xg,
                                                const unsigned short* __restrict__ wT,
                                                const float* __restrict__ bias,
                                                float* __restrict__ out) {
    const int lane = threadIdx.x & 31;
    const int wave = threadIdx.x >> 5;
    const int half = lane >> 4;
    const int l16  = lane & 15;
    const int n    = blockIdx.x * 4 + wave;

    v8f acc[4][4];
#pragma unroll
    for (int r = 0; r < 4; ++r)
#pragma unroll
        for (int c = 0; c < 4; ++c) acc[r][c] = v8f_zero();

    const unsigned short* xgn = xg + (size_t)n * PB * (PK * PC);
    const unsigned short* wtn = wT + (size_t)n * PH * (PK * PC);

#pragma unroll
    for (int kc = 0; kc < PK; ++kc) {
        BfFrag a[4], b[4];
#pragma unroll
        for (int br = 0; br < 4; ++br) {
            const unsigned short* p =
                xgn + (size_t)(br * 16 + l16) * 96 + kc * 32 + half * 8;
            a[br].q[0] = *reinterpret_cast<const v4i*>(p);
            a[br].q[1] = *reinterpret_cast<const v4i*>(p + 16);
        }
#pragma unroll
        for (int oc = 0; oc < 4; ++oc) {
            const unsigned short* p =
                wtn + (size_t)(oc * 16 + l16) * 96 + kc * 32 + half * 8;
            b[oc].q[0] = *reinterpret_cast<const v4i*>(p);
            b[oc].q[1] = *reinterpret_cast<const v4i*>(p + 16);
        }
#pragma unroll
        for (int br = 0; br < 4; ++br)
#pragma unroll
            for (int oc = 0; oc < 4; ++oc)
                acc[br][oc] = __builtin_amdgcn_wmma_f32_16x16x32_bf16(
                    false, a[br].bf, false, b[oc].bf, (short)0, acc[br][oc],
                    false, false);
    }

#pragma unroll
    for (int oc = 0; oc < 4; ++oc) {
        int o = oc * 16 + l16;
        float bv = bias[(size_t)n * PH + o];
#pragma unroll
        for (int br = 0; br < 4; ++br) {
#pragma unroll
            for (int r = 0; r < 8; ++r) {
                int bb = br * 16 + r + 8 * half;
                out[((size_t)bb * PN + n) * PH + o] = acc[br][oc][r] + bv;
            }
        }
    }
}

// ---------------------------------------------------------------------------
// Host launcher
// ---------------------------------------------------------------------------
extern "C" void kernel_launch(void* const* d_in, const int* in_sizes, int n_in,
                              void* d_out, int out_size, void* d_ws, size_t ws_size,
                              hipStream_t stream) {
    const float* x    = (const float*)d_in[0];  // (B,N,C)
    const float* emb  = (const float*)d_in[1];  // (N,D)
    const float* As   = (const float*)d_in[2];  // (K,N,N)
    const float* Wp   = (const float*)d_in[3];  // (D,K,C,H)
    const float* bp   = (const float*)d_in[4];  // (D,H)
    float* out        = (float*)d_out;          // (B,N,H)

    char* ws = (char*)d_ws;
    const size_t off_As   = 0;                                   // 100,663,296 B
    const size_t off_xcm  = off_As  + (size_t)PK * PN * PN * 2;  //  16,777,216 B
    const size_t off_wT   = off_xcm + (size_t)NCOLS * PN * 2;    //  50,331,648 B
    const size_t off_bias = off_wT  + (size_t)PN * PH * PK * PC * 2; // 1,048,576 B
    const size_t off_xg   = off_bias + (size_t)PN * PH * 4;      //  50,331,648 B

    unsigned short* Abf  = (unsigned short*)(ws + off_As);
    unsigned short* xcm  = (unsigned short*)(ws + off_xcm);
    unsigned short* wT   = (unsigned short*)(ws + off_wT);
    float*          bias = (float*)(ws + off_bias);
    unsigned short* xg   = (unsigned short*)(ws + off_xg);

    // --- prep: precision conversion + layout transposes + adaptive weights ---
    {
        long nAs = (long)PK * PN * PN;                        // 50,331,648
        k_cvt_as<<<dim3((unsigned)((nAs + 255) / 256)), dim3(256), 0, stream>>>(As, Abf, nAs);

        long nX = (long)NCOLS * PN;                           // 8,388,608
        k_build_xcm<<<dim3((unsigned)((nX + 255) / 256)), dim3(256), 0, stream>>>(x, xcm);

        long nW = (long)PN * PH * PK * PC;                    // 25,165,824
        k_build_wt<<<dim3((unsigned)((nW + 255) / 256)), dim3(256), 0, stream>>>(emb, Wp, wT);

        long nBi = (long)PN * PH;                             // 262,144
        k_build_bias<<<dim3((unsigned)((nBi + 255) / 256)), dim3(256), 0, stream>>>(emb, bp, bias);
    }

    // --- main diffusion GEMM: grid (N/128, NCOLS/256, K) of 8-wave WGs ---
    k_diffusion<<<dim3(PN / 128, NCOLS / 256, PK), dim3(256), 0, stream>>>(Abf, xcm, xg);

    // --- per-node output GEMM + bias: one wave per node ---
    k_output<<<dim3(PN / 4), dim3(128), 0, stream>>>(xg, wT, bias, out);
}
